// _TanhMultiHeadAttention_3384434229463
// MI455X (gfx1250) — matrix operand order repaired and checked by reading, not compiled, not run on hardware
//
#include <hip/hip_runtime.h>
#include <hip/hip_bf16.h>

typedef __attribute__((ext_vector_type(16))) __bf16       v16bf;
typedef __attribute__((ext_vector_type(8)))  float        v8f;
typedef __attribute__((ext_vector_type(4)))  unsigned int v4u;

union BfVec { v16bf v; v4u u[2]; };

#define BATCH 4
#define LSEQ  1024
#define EMB   1024
#define NH    16
#define HD    64
#define SCALE 0.125f
#define TINYF 1.175494350822287508e-38f

// ---------- sortable-float encoding for atomic max ----------
__device__ __forceinline__ unsigned int fenc(float f) {
  unsigned int u = __float_as_uint(f);
  return (u & 0x80000000u) ? ~u : (u | 0x80000000u);
}
__device__ __forceinline__ float fdec(unsigned int u) {
  unsigned int v = (u & 0x80000000u) ? (u & 0x7fffffffu) : ~u;
  return __uint_as_float(v);
}

// ---------- block reductions (LDS trees) ----------
__device__ __forceinline__ float blockReduceMax(float v, float* red) {
  const int tid = threadIdx.x;
  __syncthreads();
  red[tid] = v; __syncthreads();
  for (int s = 128; s > 0; s >>= 1) {
    if (tid < s) red[tid] = fmaxf(red[tid], red[tid + s]);
    __syncthreads();
  }
  return red[0];
}
__device__ __forceinline__ float blockReduceSum(float v, float* red) {
  const int tid = threadIdx.x;
  __syncthreads();
  red[tid] = v; __syncthreads();
  for (int s = 128; s > 0; s >>= 1) {
    if (tid < s) red[tid] = red[tid] + red[tid + s];
    __syncthreads();
  }
  return red[0];
}

// ---------- conversions ----------
__global__ void __launch_bounds__(256)
k_init(unsigned int* maxu, float* denom, float* gsum) {
  const int i = threadIdx.x;
  if (i < BATCH) { maxu[i] = 0u; denom[i] = 0.f; gsum[i] = 0.f; }
}

__global__ void __launch_bounds__(256)
k_cvt(const float* __restrict__ src, __bf16* __restrict__ dst, int n) {
  const int i = blockIdx.x * 256 + threadIdx.x;
  if (i < n) dst[i] = (__bf16)src[i];
}

// 1024x1024 transpose + bf16 convert: dst[(c,K) row-major] = src[r][c]
__global__ void __launch_bounds__(256)
k_cvt_t(const float* __restrict__ src, __bf16* __restrict__ dst) {
  const int i = blockIdx.x * 256 + threadIdx.x;   // < 1048576
  const int r = i >> 10, c = i & 1023;
  dst[((size_t)c << 10) + r] = (__bf16)src[i];
}

// ---------- generic bf16 WMMA GEMM: C(MxN) = A(MxK,row) * Bt(NxK,row)^T ----------
// mode 0: f32 store to outF[m*N+n]
// mode 1: bf16 head store   outH[((b*NH+h)*LSEQ+l)*HD+d]  (m=b*L+l, n=h*64+d)
// mode 2: bf16 head store T outH[((b*NH+h)*HD+d)*LSEQ+l]
__global__ void __launch_bounds__(256)
k_gemm_bf16(const __bf16* __restrict__ A, const __bf16* __restrict__ Bt,
            int M, int N, int K, int mode,
            float* __restrict__ outF, __bf16* __restrict__ outH) {
  const int tid  = threadIdx.x;
  const int lane = tid & 31;
  const int wave = tid >> 5;
  const int tilesN = N >> 4;
  const int gt = blockIdx.x * 8 + wave;
  if (gt >= (M >> 4) * tilesN) return;
  const int m0 = (gt / tilesN) << 4;
  const int n0 = (gt % tilesN) << 4;

  const int rowA = m0 + (lane & 15);
  const int k0a  = (lane & 16) ? 8 : 0;     // A: chunks at k0a and k0a+16
  const int rowB = n0 + (lane & 15);
  const int k0b  = (lane & 16) ? 8 : 0;     // B: chunks at k0b and k0b+16
  const __bf16* aBase = A  + (size_t)rowA * K + k0a;
  const __bf16* bBase = Bt + (size_t)rowB * K + k0b;

  v8f acc = {0.f, 0.f, 0.f, 0.f, 0.f, 0.f, 0.f, 0.f};
  for (int kk = 0; kk < K; kk += 32) {
    BfVec av, bv;
    av.u[0] = *(const v4u*)(aBase + kk);
    av.u[1] = *(const v4u*)(aBase + kk + 16);
    bv.u[0] = *(const v4u*)(bBase + kk);
    bv.u[1] = *(const v4u*)(bBase + kk + 16);
    acc = __builtin_amdgcn_wmma_f32_16x16x32_bf16(false, av.v, false, bv.v,
                                                  (short)0, acc, false, false);
  }

  const int n     = n0 + (lane & 15);
  const int rbase = (lane & 16) ? 8 : 0;
  #pragma unroll
  for (int r = 0; r < 8; ++r) {
    const int m = m0 + rbase + r;
    const float val = acc[r];
    if (mode == 0) {
      outF[(size_t)m * N + n] = val;
    } else {
      const int b = m >> 10, l = m & 1023;
      const int h = n >> 6,  d = n & 63;
      if (mode == 1)
        outH[(((size_t)(b * NH + h)) * LSEQ + l) * HD + d] = (__bf16)val;
      else
        outH[(((size_t)(b * NH + h)) * HD + d) * LSEQ + l] = (__bf16)val;
    }
  }
}

// ---------- scores: S[bh] = tanh(scale * Q Kt) + per-batch atomic max over valid keys ----------
__global__ void __launch_bounds__(256)
k_scores(const __bf16* __restrict__ Q, const __bf16* __restrict__ Kb,
         const unsigned char* __restrict__ mask,
         float* __restrict__ Wbuf, unsigned int* __restrict__ maxu) {
  __shared__ float red[256];
  const int tid  = threadIdx.x;
  const int lane = tid & 31;
  const int wave = tid >> 5;
  const int bh   = blockIdx.x >> 9;                       // 512 blocks per (b,h)
  const int tIn  = ((blockIdx.x & 511) << 3) + wave;      // 0..4095
  const int m0   = (tIn >> 6) << 4;                       // query tile
  const int n0   = (tIn & 63) << 4;                       // key tile
  const int b    = bh >> 4;

  const __bf16* A  = Q  + (size_t)bh * LSEQ * HD;
  const __bf16* Bt = Kb + (size_t)bh * LSEQ * HD;
  const int rowA = m0 + (lane & 15);
  const int k0a  = (lane & 16) ? 8 : 0;
  const int rowB = n0 + (lane & 15);
  const int k0b  = (lane & 16) ? 8 : 0;
  const __bf16* aBase = A  + (size_t)rowA * HD + k0a;
  const __bf16* bBase = Bt + (size_t)rowB * HD + k0b;

  v8f acc = {0.f, 0.f, 0.f, 0.f, 0.f, 0.f, 0.f, 0.f};
  #pragma unroll
  for (int kk = 0; kk < HD; kk += 32) {
    BfVec av, bv;
    av.u[0] = *(const v4u*)(aBase + kk);
    av.u[1] = *(const v4u*)(aBase + kk + 16);
    bv.u[0] = *(const v4u*)(bBase + kk);
    bv.u[1] = *(const v4u*)(bBase + kk + 16);
    acc = __builtin_amdgcn_wmma_f32_16x16x32_bf16(false, av.v, false, bv.v,
                                                  (short)0, acc, false, false);
  }

  const int nn    = n0 + (lane & 15);                     // key index
  const int rbase = (lane & 16) ? 8 : 0;
  const int kmk   = mask[(b << 10) + nn];
  float lmax = -3.402823466e38f;
  #pragma unroll
  for (int r = 0; r < 8; ++r) {
    const int m = m0 + rbase + r;                         // query index
    const float s = tanhf(acc[r] * SCALE);
    Wbuf[((size_t)bh * LSEQ + m) * LSEQ + nn] = s;        // raw tanh score
    if (kmk) lmax = fmaxf(lmax, s);                       // max over valid keys, all queries
  }
  const float bm = blockReduceMax(lmax, red);
  if (tid == 0) atomicMax(&maxu[b], fenc(bm));
}

// ---------- exp pass (in place) + per-batch denom ----------
__global__ void __launch_bounds__(256)
k_exppass(float* __restrict__ W, const unsigned char* __restrict__ mask,
          const unsigned int* __restrict__ maxu, float* __restrict__ denom) {
  __shared__ float red[256];
  const int tid = threadIdx.x;
  const int row = blockIdx.x;             // row = bh*1024 + q  (65536 rows)
  const int q   = row & 1023;
  const int b   = row >> 14;
  const float m = fdec(maxu[b]);
  const int kmq = mask[(b << 10) + q];
  float* p = W + ((size_t)row << 10) + (tid << 2);
  float4 s = *(float4*)p;
  const int kb = tid << 2;
  float4 e;
  e.x = (kmq && mask[(b << 10) + kb + 0]) ? __expf(s.x - m) : 0.f;
  e.y = (kmq && mask[(b << 10) + kb + 1]) ? __expf(s.y - m) : 0.f;
  e.z = (kmq && mask[(b << 10) + kb + 2]) ? __expf(s.z - m) : 0.f;
  e.w = (kmq && mask[(b << 10) + kb + 3]) ? __expf(s.w - m) : 0.f;
  *(float4*)p = e;
  const float bs = blockReduceSum(e.x + e.y + e.z + e.w, red);
  if (tid == 0) atomicAdd(&denom[b], bs);
}

// ---------- row softmax of G -> act ----------
__global__ void __launch_bounds__(256)
k_gsoftmax(const float* __restrict__ G, float* __restrict__ act) {
  __shared__ float red[256];
  const int tid = threadIdx.x;
  const int row = blockIdx.x;
  const float* g = G + ((size_t)row << 10);
  float lm = -3.402823466e38f;
  for (int i = tid; i < 1024; i += 256) lm = fmaxf(lm, g[i]);
  const float bm = blockReduceMax(lm, red);
  float ls = 0.f;
  for (int i = tid; i < 1024; i += 256) ls += __expf(g[i] - bm);
  const float bs  = blockReduceSum(ls, red);
  const float inv = 1.0f / bs;
  for (int i = tid; i < 1024; i += 256)
    act[((size_t)row << 10) + i] = __expf(g[i] - bm) * inv;
}

// ---------- per-batch masked sum of act ----------
__global__ void __launch_bounds__(256)
k_gsum(const float* __restrict__ act, const unsigned char* __restrict__ mask,
       float* __restrict__ gsum) {
  __shared__ float red[256];
  const int tid = threadIdx.x;
  const int q = blockIdx.x & 1023;
  const int b = blockIdx.x >> 10;
  const int kmq = mask[(b << 10) + q];
  float ls = 0.f;
  if (kmq)
    for (int i = tid; i < 1024; i += 256)
      ls += mask[(b << 10) + i] ? act[((size_t)q << 10) + i] : 0.f;
  const float bs = blockReduceSum(ls, red);
  if (tid == 0) atomicAdd(&gsum[b], bs);
}

// ---------- final weights = 0.5*ex/denom + 0.5*gnorm (in place) ----------
__global__ void __launch_bounds__(256)
k_mix(float* __restrict__ W, const float* __restrict__ act,
      const unsigned char* __restrict__ mask,
      const float* __restrict__ denom, const float* __restrict__ gsum) {
  const int tid = threadIdx.x;
  const int row = blockIdx.x;             // bh*1024 + q
  const int q   = row & 1023;
  const int b   = row >> 14;
  const int kmq = mask[(b << 10) + q];
  const float dn = 1.0f / fmaxf(denom[b], TINYF);
  const float gs = 1.0f / fmaxf(gsum[b], TINYF);
  float* p = W + ((size_t)row << 10) + (tid << 2);
  float4 ex = *(float4*)p;
  const float4 av = *(const float4*)(act + ((size_t)q << 10) + (tid << 2));
  const int kb = tid << 2;
  float4 o;
  o.x = 0.5f * ex.x * dn + ((kmq && mask[(b<<10)+kb+0]) ? 0.5f * av.x * gs : 0.f);
  o.y = 0.5f * ex.y * dn + ((kmq && mask[(b<<10)+kb+1]) ? 0.5f * av.y * gs : 0.f);
  o.z = 0.5f * ex.z * dn + ((kmq && mask[(b<<10)+kb+2]) ? 0.5f * av.z * gs : 0.f);
  o.w = 0.5f * ex.w * dn + ((kmq && mask[(b<<10)+kb+3]) ? 0.5f * av.w * gs : 0.f);
  *(float4*)p = o;
}

// ---------- attended = weights @ V (A: f32 -> bf16 on the fly; B = Vt (d,l)) ----------
__global__ void __launch_bounds__(256)
k_attend(const float* __restrict__ Wgt, const __bf16* __restrict__ Vt,
         const unsigned char* __restrict__ mask, __bf16* __restrict__ att) {
  const int tid  = threadIdx.x;
  const int lane = tid & 31;
  const int wave = tid >> 5;
  const int bh   = blockIdx.x >> 5;                    // 32 blocks per (b,h)
  const int tIn  = ((blockIdx.x & 31) << 3) + wave;    // 0..255
  const int m0   = (tIn >> 2) << 4;                    // query tile (64 tiles)
  const int n0   = (tIn & 3) << 4;                     // d tile (4 tiles)

  const float*  A  = Wgt + (size_t)bh * LSEQ * LSEQ;
  const __bf16* Bt = Vt  + (size_t)bh * HD * LSEQ;
  const int rowA = m0 + (lane & 15);
  const int k0a  = (lane & 16) ? 8 : 0;
  const int rowB = n0 + (lane & 15);
  const int k0b  = (lane & 16) ? 8 : 0;
  const float*  aBase = A  + (size_t)rowA * LSEQ + k0a;
  const __bf16* bBase = Bt + (size_t)rowB * LSEQ + k0b;

  v8f acc = {0.f, 0.f, 0.f, 0.f, 0.f, 0.f, 0.f, 0.f};
  for (int kk = 0; kk < LSEQ; kk += 32) {
    const float4 f0 = *(const float4*)(aBase + kk);
    const float4 f1 = *(const float4*)(aBase + kk + 4);
    const float4 f2 = *(const float4*)(aBase + kk + 16);
    const float4 f3 = *(const float4*)(aBase + kk + 20);
    v16bf av;
    av[0]=(__bf16)f0.x;  av[1]=(__bf16)f0.y;  av[2]=(__bf16)f0.z;  av[3]=(__bf16)f0.w;
    av[4]=(__bf16)f1.x;  av[5]=(__bf16)f1.y;  av[6]=(__bf16)f1.z;  av[7]=(__bf16)f1.w;
    av[8]=(__bf16)f2.x;  av[9]=(__bf16)f2.y;  av[10]=(__bf16)f2.z; av[11]=(__bf16)f2.w;
    av[12]=(__bf16)f3.x; av[13]=(__bf16)f3.y; av[14]=(__bf16)f3.z; av[15]=(__bf16)f3.w;
    BfVec bv;
    bv.u[0] = *(const v4u*)(bBase + kk);
    bv.u[1] = *(const v4u*)(bBase + kk + 16);
    acc = __builtin_amdgcn_wmma_f32_16x16x32_bf16(false, av, false, bv.v,
                                                  (short)0, acc, false, false);
  }

  const int d     = n0 + (lane & 15);
  const int rbase = (lane & 16) ? 8 : 0;
  const int b = bh >> 4, h = bh & 15;
  #pragma unroll
  for (int r = 0; r < 8; ++r) {
    const int l = m0 + rbase + r;
    const int kmq = mask[(b << 10) + l];
    const float v = kmq ? acc[r] : 0.f;
    att[(((size_t)(b << 10)) + l) * EMB + (h << 6) + d] = (__bf16)v;
  }
}

// ---------- host launcher ----------
extern "C" void kernel_launch(void* const* d_in, const int* in_sizes, int n_in,
                              void* d_out, int out_size, void* d_ws, size_t ws_size,
                              hipStream_t stream) {
  const float*         x    = (const float*)d_in[0];          // (B,L,E)
  const unsigned char* mask = (const unsigned char*)d_in[1];  // (B,L) bool
  const float* Wq = (const float*)d_in[2];
  const float* Wk = (const float*)d_in[3];
  const float* Wv = (const float*)d_in[4];
  const float* Wo = (const float*)d_in[5];
  const float* G  = (const float*)d_in[6];

  float* out  = (float*)d_out;                                // (B,L,E)
  float* wout = out + (size_t)BATCH * LSEQ * EMB;             // (B,H,L,L) weights

  char* ws = (char*)d_ws;
  __bf16* XBF = (__bf16*)(ws + 0);               // 4096x1024 bf16
  __bf16* WQT = (__bf16*)(ws + 8388608);         // 1024x1024 bf16 (transposed)
  __bf16* WKT = (__bf16*)(ws + 10485760);
  __bf16* WVT = (__bf16*)(ws + 12582912);
  __bf16* WOT = (__bf16*)(ws + 14680064);
  __bf16* QBF = (__bf16*)(ws + 16777216);        // (b,h,l,d) bf16
  __bf16* KBF = (__bf16*)(ws + 25165824);        // (b,h,l,d) bf16
  __bf16* VTB = (__bf16*)(ws + 33554432);        // (b,h,d,l) bf16
  float*  ACT = (float*)(ws + 41943040);         // softmax(G) f32
  __bf16* ATT = (__bf16*)(ws + 46137344);        // (B*L, E) bf16
  unsigned int* MAXU  = (unsigned int*)(ws + 54525952);
  float*        DENOM = (float*)(ws + 54525952 + 16);
  float*        GSUM  = (float*)(ws + 54525952 + 32);

  hipLaunchKernelGGL(k_init, dim3(1), dim3(256), 0, stream, MAXU, DENOM, GSUM);
  hipLaunchKernelGGL(k_cvt, dim3(16384), dim3(256), 0, stream, x, XBF, BATCH*LSEQ*EMB);
  hipLaunchKernelGGL(k_cvt_t, dim3(4096), dim3(256), 0, stream, Wq, WQT);
  hipLaunchKernelGGL(k_cvt_t, dim3(4096), dim3(256), 0, stream, Wk, WKT);
  hipLaunchKernelGGL(k_cvt_t, dim3(4096), dim3(256), 0, stream, Wv, WVT);
  hipLaunchKernelGGL(k_cvt_t, dim3(4096), dim3(256), 0, stream, Wo, WOT);

  // Q/K (b,h,l,d) and V transposed (b,h,d,l)
  hipLaunchKernelGGL(k_gemm_bf16, dim3(2048), dim3(256), 0, stream,
                     XBF, WQT, BATCH*LSEQ, EMB, EMB, 1, (float*)nullptr, QBF);
  hipLaunchKernelGGL(k_gemm_bf16, dim3(2048), dim3(256), 0, stream,
                     XBF, WKT, BATCH*LSEQ, EMB, EMB, 1, (float*)nullptr, KBF);
  hipLaunchKernelGGL(k_gemm_bf16, dim3(2048), dim3(256), 0, stream,
                     XBF, WVT, BATCH*LSEQ, EMB, EMB, 2, (float*)nullptr, VTB);

  hipLaunchKernelGGL(k_scores,   dim3(32768), dim3(256), 0, stream, QBF, KBF, mask, wout, MAXU);
  hipLaunchKernelGGL(k_exppass,  dim3(65536), dim3(256), 0, stream, wout, mask, MAXU, DENOM);
  hipLaunchKernelGGL(k_gsoftmax, dim3(1024),  dim3(256), 0, stream, G, ACT);
  hipLaunchKernelGGL(k_gsum,     dim3(4096),  dim3(256), 0, stream, ACT, mask, GSUM);
  hipLaunchKernelGGL(k_mix,      dim3(65536), dim3(256), 0, stream, wout, ACT, mask, DENOM, GSUM);

  hipLaunchKernelGGL(k_attend, dim3(2048), dim3(256), 0, stream, wout, VTB, mask, ATT);
  hipLaunchKernelGGL(k_gemm_bf16, dim3(2048), dim3(256), 0, stream,
                     ATT, WOT, BATCH*LSEQ, EMB, EMB, 0, out, (__bf16*)nullptr);
}
